// BidirectionalS4DBlock_64175401337567
// MI455X (gfx1250) — matrix operand order repaired and checked by reading, not compiled, not run on hardware
//
#include <hip/hip_runtime.h>
#include <hip/hip_bf16.h>

// ---------------------------------------------------------------------------
// BidirectionalS4DBlock for MI455X (gfx1250, wave32, WMMA + TDM)
//
//   1. convert+transpose weights fp32 -> bf16
//   2. S4D discretization params per (dir,d,n)
//   3. film = silu(emb) @ film_W + film_b
//   4. h = LayerNorm(x)
//   5. h_t = transpose(h)  (B,L,d)->(B,d,L)
//   6. bidirectional modal scan -> ybi bf16 (B,L,2d)
//   7. GEMM1 (WMMA bf16, TDM double-buffer): y = ybi@bidir_W + b + h*D, FiLM
//   8. GEMM2 (WMMA bf16, TDM): t = silu(y@gate_W+gb) * (y@up_W+ub)
//   9. GEMM3 (WMMA bf16, TDM): out = x + t@down_W + db
//
// Workspace requirement: ~210 MB.
// ---------------------------------------------------------------------------

typedef __bf16 bf16_t;
typedef __attribute__((ext_vector_type(16))) __bf16 v16bf;
typedef __attribute__((ext_vector_type(8)))  __bf16 v8bf;
typedef __attribute__((ext_vector_type(8)))  float  v8f;
typedef __attribute__((ext_vector_type(4)))  unsigned int u32x4;
typedef __attribute__((ext_vector_type(8)))  int i32x8;
typedef __attribute__((ext_vector_type(4)))  int i32x4;

#define D_MODEL 1024
#define TWO_D   2048
#define BATCH   8
#define SEQ_L   2048
#define M_TOK   (BATCH * SEQ_L)   // 16384
#define EMB_E   512
#define NMODES  32

#if __has_builtin(__builtin_amdgcn_tensor_load_to_lds) && \
    __has_builtin(__builtin_amdgcn_s_wait_tensorcnt)
#define HAVE_TDM 1
#else
#define HAVE_TDM 0
#endif

// ======================= small prep kernels =================================

__global__ void __launch_bounds__(256)
convT_kernel(const float* __restrict__ src, bf16_t* __restrict__ dst, int R, int C) {
  __shared__ float tile[32][33];
  int c0 = blockIdx.x * 32, r0 = blockIdx.y * 32;
  int tx = threadIdx.x, ty = threadIdx.y;            // block (32,8)
  for (int i = ty; i < 32; i += 8)
    tile[i][tx] = src[(size_t)(r0 + i) * C + c0 + tx];
  __syncthreads();
  for (int i = ty; i < 32; i += 8)
    dst[(size_t)(c0 + i) * R + r0 + tx] = (bf16_t)tile[tx][i];
}

__global__ void __launch_bounds__(256)
s4d_params_kernel(const float* __restrict__ log_dt, const float* __restrict__ log_A_real,
                  const float* __restrict__ A_imag, const float* __restrict__ C_re,
                  const float* __restrict__ C_im, float4* __restrict__ out) {
  int i = blockIdx.x * 256 + threadIdx.x;
  if (i >= D_MODEL * NMODES) return;
  int dch = i >> 5;
  float dt = __expf(log_dt[dch]);
  float Ar = -__expf(log_A_real[i]);
  float Ai = A_imag[i];
  float er = Ar * dt, ei = Ai * dt;
  float m  = __expf(er);
  float ar = m * __cosf(ei), ai = m * __sinf(ei);
  float nr = ar - 1.f, ni = ai;
  float inv = 1.f / (Ar * Ar + Ai * Ai);
  float qr = (nr * Ar + ni * Ai) * inv;
  float qi = (ni * Ar - nr * Ai) * inv;
  float cr = C_re[i], ci = C_im[i];
  out[i] = make_float4(ar, ai, 2.f * (cr * qr - ci * qi), 2.f * (cr * qi + ci * qr));
}

__global__ void __launch_bounds__(256)
film_kernel(const float* __restrict__ emb, const float* __restrict__ W,
            const float* __restrict__ bvec, float* __restrict__ film) {
  int j = blockIdx.x * 256 + threadIdx.x;
  if (j >= BATCH * TWO_D) return;
  int b = j >> 11, c = j & (TWO_D - 1);
  float s = 0.f;
  for (int k = 0; k < EMB_E; ++k) {
    float e  = emb[b * EMB_E + k];
    float se = e / (1.f + __expf(-e));
    s += se * W[(size_t)k * TWO_D + c];
  }
  film[j] = s + bvec[c];
}

__global__ void __launch_bounds__(256)
ln_kernel(const float* __restrict__ x, const float* __restrict__ g,
          const float* __restrict__ bb, float* __restrict__ h) {
  int row = blockIdx.x;
  const float* xr = x + (size_t)row * D_MODEL;
  float s = 0.f, s2 = 0.f;
  for (int i = threadIdx.x; i < D_MODEL; i += 256) {
    float v = xr[i]; s += v; s2 += v * v;
  }
  for (int o = 16; o; o >>= 1) { s += __shfl_xor(s, o); s2 += __shfl_xor(s2, o); }
  __shared__ float red[18];
  if ((threadIdx.x & 31) == 0) {
    red[(threadIdx.x >> 5) * 2] = s; red[(threadIdx.x >> 5) * 2 + 1] = s2;
  }
  __syncthreads();
  if (threadIdx.x == 0) {
    float a = 0.f, c = 0.f;
    for (int w = 0; w < 8; ++w) { a += red[2 * w]; c += red[2 * w + 1]; }
    float mu = a * (1.f / D_MODEL);
    red[16] = mu; red[17] = c * (1.f / D_MODEL) - mu * mu;
  }
  __syncthreads();
  float mu = red[16];
  float inv = rsqrtf(red[17] + 1e-5f);
  for (int i = threadIdx.x; i < D_MODEL; i += 256)
    h[(size_t)row * D_MODEL + i] = (xr[i] - mu) * inv * g[i] + bb[i];
}

__global__ void __launch_bounds__(256)
transpose_kernel(const float* __restrict__ src, float* __restrict__ dst) {
  __shared__ float tile[32][33];
  int l0 = blockIdx.x * 32, d0 = blockIdx.y * 32, b = blockIdx.z;
  int tx = threadIdx.x, ty = threadIdx.y;                // block (32,8)
  const float* s = src + (size_t)b * SEQ_L * D_MODEL;
  for (int i = ty; i < 32; i += 8)
    tile[i][tx] = s[(size_t)(l0 + i) * D_MODEL + d0 + tx];
  __syncthreads();
  float* dp = dst + (size_t)b * D_MODEL * SEQ_L;
  for (int i = ty; i < 32; i += 8)
    dp[(size_t)(d0 + i) * SEQ_L + l0 + tx] = tile[tx][i];
}

// ======================= S4D modal scan =====================================
__global__ void __launch_bounds__(128)
scan_kernel(const float* __restrict__ h_t, const float4* __restrict__ params,
            bf16_t* __restrict__ ybi) {
  int blk = blockIdx.x;                  // 2 * 8 * 64 = 1024 blocks
  int dir = blk >> 9;
  int rem = blk & 511;
  int b   = rem >> 6;
  int d0  = (rem & 63) << 4;
  int tid  = threadIdx.x;
  int unit = tid >> 3;                   // 0..15 -> d = d0 + unit
  int sub  = tid & 7;                    // 0..7  -> modes sub*4 .. sub*4+3
  int d = d0 + unit;

  const float4* pp = params + ((size_t)dir * D_MODEL + d) * NMODES + sub * 4;
  float4 p0 = pp[0], p1 = pp[1], p2 = pp[2], p3 = pp[3];
  float sr0 = 0.f, si0 = 0.f, sr1 = 0.f, si1 = 0.f;
  float sr2 = 0.f, si2 = 0.f, sr3 = 0.f, si3 = 0.f;

  __shared__ float htile[16][33];
  __shared__ float ytile[16][32];
  const float* hb = h_t + (size_t)b * D_MODEL * SEQ_L;
  bf16_t* yb = ybi + (size_t)b * SEQ_L * TWO_D + dir * D_MODEL;

  for (int l0 = 0; l0 < SEQ_L; l0 += 32) {
#pragma unroll
    for (int i = 0; i < 4; ++i) {
      int e = i * 128 + tid;
      int r = e >> 5, c = e & 31;
      int pos = dir ? (SEQ_L - 1 - (l0 + c)) : (l0 + c);
      htile[r][c] = hb[(size_t)(d0 + r) * SEQ_L + pos];
    }
    __syncthreads();
#pragma unroll 4
    for (int j = 0; j < 32; ++j) {
      float hv = htile[unit][j];
      float t;
      t = fmaf(p0.x, sr0, fmaf(-p0.y, si0, hv));
      si0 = fmaf(p0.x, si0, p0.y * sr0); sr0 = t;
      t = fmaf(p1.x, sr1, fmaf(-p1.y, si1, hv));
      si1 = fmaf(p1.x, si1, p1.y * sr1); sr1 = t;
      t = fmaf(p2.x, sr2, fmaf(-p2.y, si2, hv));
      si2 = fmaf(p2.x, si2, p2.y * sr2); sr2 = t;
      t = fmaf(p3.x, sr3, fmaf(-p3.y, si3, hv));
      si3 = fmaf(p3.x, si3, p3.y * sr3); sr3 = t;
      float r = fmaf(p0.z, sr0, -p0.w * si0) + fmaf(p1.z, sr1, -p1.w * si1) +
                fmaf(p2.z, sr2, -p2.w * si2) + fmaf(p3.z, sr3, -p3.w * si3);
      r += __shfl_xor(r, 1);
      r += __shfl_xor(r, 2);
      r += __shfl_xor(r, 4);
      if (sub == 0) ytile[unit][j] = r;
    }
    __syncthreads();
#pragma unroll
    for (int i = 0; i < 4; ++i) {
      int e = i * 128 + tid;
      int lr = e >> 4, c = e & 15;
      int pos = dir ? (SEQ_L - 1 - (l0 + lr)) : (l0 + lr);
      yb[(size_t)pos * TWO_D + d0 + c] = (bf16_t)ytile[c][lr];
    }
    __syncthreads();
  }
}

// ======================= WMMA GEMM building blocks ==========================
// LDS tiles: rows x 32 bf16, row stride 40 elems (80B = 64B data + 16B pad)
#define LDS_STRIDE 40

// sync fallback tile loader
template <int ROWS>
__device__ __forceinline__ void load_tile_sync(const bf16_t* __restrict__ g, int ld,
                                               int row0, int k0, bf16_t* lds, int tid) {
#pragma unroll
  for (int i = 0; i < ROWS / 64; ++i) {
    int seg = tid + i * 256;
    int r  = seg >> 2;
    int ks = (seg & 3) << 3;
    *(uint4*)(lds + r * LDS_STRIDE + ks) =
        *(const uint4*)(g + (size_t)(row0 + r) * ld + k0 + ks);
  }
}

#if HAVE_TDM
// One TDM descriptor: 2D tile (tile_rows x 32 bf16) from row-major matrix
// (ld elems/row). LDS padding: +16B after every 64B row -> stride 40 elems.
__device__ __forceinline__ void tdm_load_tile(const bf16_t* gtile, unsigned lds_off,
                                              unsigned ld, unsigned rows_total,
                                              unsigned tile_rows) {
  unsigned long long ga = (unsigned long long)(uintptr_t)(const void*)gtile;
  u32x4 g0 = {
      1u,                                        // count=1, user mode, no gather
      lds_off,                                   // lds_addr [63:32]
      (unsigned)(ga & 0xffffffffu),              // global_addr [95:64]
      (unsigned)((ga >> 32) & 0x01ffffffu) | (2u << 30)  // addr hi | type=2
  };
  i32x8 g1 = {
      (int)((1u << 16) | (1u << 20) | (3u << 22) | (3u << 25)),
      // ^ data_size=2B, pad_enable, pad_interval=3 (64B), pad_amount=3 (16B)
      (int)((ld & 0xffffu) << 16),               // abar=0 | tensor_dim0 lo
      (int)((ld >> 16) | ((rows_total & 0xffffu) << 16)),  // dim0 hi | dim1 lo
      (int)((rows_total >> 16) | (32u << 16)),   // dim1 hi | tile_dim0=32
      (int)(tile_rows),                          // tile_dim1 | tile_dim2=0
      (int)ld,                                   // tensor_dim0_stride lo
      0, 0
  };
  i32x4 z4 = {0, 0, 0, 0};
#if __clang_major__ >= 23
  i32x8 z8 = {0, 0, 0, 0, 0, 0, 0, 0};
  __builtin_amdgcn_tensor_load_to_lds(g0, g1, z4, z4, z8, 0);
#else
  __builtin_amdgcn_tensor_load_to_lds(g0, g1, z4, z4, 0);
#endif
}
__device__ __forceinline__ unsigned lds_off_of(const void* p) {
  return (unsigned)(unsigned long long)(uintptr_t)p;   // low 32 bits = LDS offset
}
#endif

// A fragment (16x32 bf16): elems 0..7 = K base..base+7, 8..15 = K base+16..+23
__device__ __forceinline__ v16bf load_fragA(const bf16_t* lds, int rowBase, int lane) {
  const bf16_t* p = lds + (rowBase + (lane & 15)) * LDS_STRIDE + ((lane >> 4) << 3);
  v8bf lo = *(const v8bf*)p;
  v8bf hi = *(const v8bf*)(p + 16);
  return __builtin_shufflevector(lo, hi, 0, 1, 2, 3, 4, 5, 6, 7,
                                 8, 9, 10, 11, 12, 13, 14, 15);
}

// B fragment (32x16 bf16) from N-major LDS tile: elems 0..7 = K base..base+7, 8..15 = K base+16..+23
__device__ __forceinline__ v16bf load_fragB(const bf16_t* lds, int rowBase, int lane) {
  const bf16_t* p = lds + (rowBase + (lane & 15)) * LDS_STRIDE + ((lane >> 4) << 3);
  v8bf lo = *(const v8bf*)p;
  v8bf hi = *(const v8bf*)(p + 16);
  return __builtin_shufflevector(lo, hi, 0, 1, 2, 3, 4, 5, 6, 7,
                                 8, 9, 10, 11, 12, 13, 14, 15);
}

#define WMMA_BF16(acc, a, b) \
  __builtin_amdgcn_wmma_f32_16x16x32_bf16(false, (a), false, (b), (short)0, (acc), false, false)

// ---- GEMM1: y = ybi @ bidir_W + bias + h*D, then FiLM; out bf16 (16384x1024)
// Block tile 128x256, 8 waves in 2x4 grid, wave tile 64x64, K-step 32.
__global__ void __launch_bounds__(256)
gemm_bidir_kernel(const bf16_t* __restrict__ A, const bf16_t* __restrict__ Wt,
                  const float* __restrict__ bias, const float* __restrict__ h,
                  const float* __restrict__ Dvec, const float* __restrict__ film,
                  bf16_t* __restrict__ out) {
  __shared__ bf16_t As[2][128 * LDS_STRIDE];
  __shared__ bf16_t Bs[2][256 * LDS_STRIDE];
  int tid = threadIdx.x, lane = tid & 31, wave = tid >> 5;
  int wm = wave & 1, wn = wave >> 1;                 // 2 x 4 wave grid
  int m0 = blockIdx.x * 128, n0 = blockIdx.y * 256;
  const int NK = TWO_D / 32;
  v8f acc[4][4] = {};
#if HAVE_TDM
  if (wave == 0) {
    tdm_load_tile(A + (size_t)m0 * TWO_D, lds_off_of(As[0]), TWO_D, M_TOK, 128);
    tdm_load_tile(Wt + (size_t)n0 * TWO_D, lds_off_of(Bs[0]), TWO_D, D_MODEL, 256);
  }
#endif
  for (int kk = 0; kk < NK; ++kk) {
    int buf = kk & 1;
#if HAVE_TDM
    if (wave == 0) {
      if (kk + 1 < NK) {
        int k1 = (kk + 1) * 32;
        tdm_load_tile(A + (size_t)m0 * TWO_D + k1, lds_off_of(As[buf ^ 1]),
                      TWO_D, M_TOK, 128);
        tdm_load_tile(Wt + (size_t)n0 * TWO_D + k1, lds_off_of(Bs[buf ^ 1]),
                      TWO_D, D_MODEL, 256);
        __builtin_amdgcn_s_wait_tensorcnt(2);
      } else {
        __builtin_amdgcn_s_wait_tensorcnt(0);
      }
    }
    __syncthreads();
#else
    load_tile_sync<128>(A, TWO_D, m0, kk * 32, As[0], tid);
    load_tile_sync<256>(Wt, TWO_D, n0, kk * 32, Bs[0], tid);
    buf = 0;
    __syncthreads();
#endif
    v16bf af[4], bf[4];
#pragma unroll
    for (int i = 0; i < 4; ++i) af[i] = load_fragA(As[buf], wm * 64 + i * 16, lane);
#pragma unroll
    for (int j = 0; j < 4; ++j) bf[j] = load_fragB(Bs[buf], wn * 64 + j * 16, lane);
#pragma unroll
    for (int i = 0; i < 4; ++i)
#pragma unroll
      for (int j = 0; j < 4; ++j)
        acc[i][j] = WMMA_BF16(acc[i][j], af[i], bf[j]);
    __syncthreads();
  }
  int hl = lane >> 4, nl = lane & 15;
#pragma unroll
  for (int i = 0; i < 4; ++i)
#pragma unroll
    for (int j = 0; j < 4; ++j) {
      int col = n0 + wn * 64 + j * 16 + nl;
      float bb = bias[col], dd = Dvec[col];
#pragma unroll
      for (int v = 0; v < 8; ++v) {
        int row = m0 + wm * 64 + i * 16 + hl * 8 + v;
        int b   = row >> 11;
        float sc = film[b * TWO_D + col];
        float sh = film[b * TWO_D + D_MODEL + col];
        float y = acc[i][j][v] + bb + h[(size_t)row * D_MODEL + col] * dd;
        y = y * (1.f + sc) + sh;
        out[(size_t)row * D_MODEL + col] = (bf16_t)y;
      }
    }
}

// ---- GEMM2: t = silu(y@gate_W+gb) * (y@up_W+ub); out bf16 (16384x2048)
// Block tile 128x128, wave tile 64x32, two B operands.
__global__ void __launch_bounds__(256)
gemm_gateup_kernel(const bf16_t* __restrict__ Y, const bf16_t* __restrict__ Gt,
                   const bf16_t* __restrict__ Ut, const float* __restrict__ gb,
                   const float* __restrict__ ub, bf16_t* __restrict__ T) {
  __shared__ bf16_t As[2][128 * LDS_STRIDE];
  __shared__ bf16_t Bg[2][128 * LDS_STRIDE];
  __shared__ bf16_t Bu[2][128 * LDS_STRIDE];
  int tid = threadIdx.x, lane = tid & 31, wave = tid >> 5;
  int wm = wave & 1, wn = wave >> 1;
  int m0 = blockIdx.x * 128, n0 = blockIdx.y * 128;
  const int NK = D_MODEL / 32;
  v8f accg[4][2] = {}, accu[4][2] = {};
#if HAVE_TDM
  if (wave == 0) {
    tdm_load_tile(Y + (size_t)m0 * D_MODEL, lds_off_of(As[0]), D_MODEL, M_TOK, 128);
    tdm_load_tile(Gt + (size_t)n0 * D_MODEL, lds_off_of(Bg[0]), D_MODEL, TWO_D, 128);
    tdm_load_tile(Ut + (size_t)n0 * D_MODEL, lds_off_of(Bu[0]), D_MODEL, TWO_D, 128);
  }
#endif
  for (int kk = 0; kk < NK; ++kk) {
    int buf = kk & 1;
#if HAVE_TDM
    if (wave == 0) {
      if (kk + 1 < NK) {
        int k1 = (kk + 1) * 32;
        tdm_load_tile(Y + (size_t)m0 * D_MODEL + k1, lds_off_of(As[buf ^ 1]),
                      D_MODEL, M_TOK, 128);
        tdm_load_tile(Gt + (size_t)n0 * D_MODEL + k1, lds_off_of(Bg[buf ^ 1]),
                      D_MODEL, TWO_D, 128);
        tdm_load_tile(Ut + (size_t)n0 * D_MODEL + k1, lds_off_of(Bu[buf ^ 1]),
                      D_MODEL, TWO_D, 128);
        __builtin_amdgcn_s_wait_tensorcnt(3);
      } else {
        __builtin_amdgcn_s_wait_tensorcnt(0);
      }
    }
    __syncthreads();
#else
    load_tile_sync<128>(Y, D_MODEL, m0, kk * 32, As[0], tid);
    load_tile_sync<128>(Gt, D_MODEL, n0, kk * 32, Bg[0], tid);
    load_tile_sync<128>(Ut, D_MODEL, n0, kk * 32, Bu[0], tid);
    buf = 0;
    __syncthreads();
#endif
    v16bf af[4], gf[2], uf[2];
#pragma unroll
    for (int i = 0; i < 4; ++i) af[i] = load_fragA(As[buf], wm * 64 + i * 16, lane);
#pragma unroll
    for (int j = 0; j < 2; ++j) {
      gf[j] = load_fragB(Bg[buf], wn * 32 + j * 16, lane);
      uf[j] = load_fragB(Bu[buf], wn * 32 + j * 16, lane);
    }
#pragma unroll
    for (int i = 0; i < 4; ++i)
#pragma unroll
      for (int j = 0; j < 2; ++j) {
        accg[i][j] = WMMA_BF16(accg[i][j], af[i], gf[j]);
        accu[i][j] = WMMA_BF16(accu[i][j], af[i], uf[j]);
      }
    __syncthreads();
  }
  int hl = lane >> 4, nl = lane & 15;
#pragma unroll
  for (int i = 0; i < 4; ++i)
#pragma unroll
    for (int j = 0; j < 2; ++j) {
      int col = n0 + wn * 32 + j * 16 + nl;
      float bgv = gb[col], buv = ub[col];
#pragma unroll
      for (int v = 0; v < 8; ++v) {
        int row = m0 + wm * 64 + i * 16 + hl * 8 + v;
        float g = accg[i][j][v] + bgv;
        float u = accu[i][j][v] + buv;
        float t = (g / (1.f + __expf(-g))) * u;
        T[(size_t)row * TWO_D + col] = (bf16_t)t;
      }
    }
}

// ---- GEMM3: out = x + t @ down_W + db; out fp32 (16384x1024)
// Block tile 128x256, wave tile 64x64.
__global__ void __launch_bounds__(256)
gemm_down_kernel(const bf16_t* __restrict__ T, const bf16_t* __restrict__ Wt,
                 const float* __restrict__ bias, const float* __restrict__ x,
                 float* __restrict__ out) {
  __shared__ bf16_t As[2][128 * LDS_STRIDE];
  __shared__ bf16_t Bs[2][256 * LDS_STRIDE];
  int tid = threadIdx.x, lane = tid & 31, wave = tid >> 5;
  int wm = wave & 1, wn = wave >> 1;
  int m0 = blockIdx.x * 128, n0 = blockIdx.y * 256;
  const int NK = TWO_D / 32;
  v8f acc[4][4] = {};
#if HAVE_TDM
  if (wave == 0) {
    tdm_load_tile(T + (size_t)m0 * TWO_D, lds_off_of(As[0]), TWO_D, M_TOK, 128);
    tdm_load_tile(Wt + (size_t)n0 * TWO_D, lds_off_of(Bs[0]), TWO_D, D_MODEL, 256);
  }
#endif
  for (int kk = 0; kk < NK; ++kk) {
    int buf = kk & 1;
#if HAVE_TDM
    if (wave == 0) {
      if (kk + 1 < NK) {
        int k1 = (kk + 1) * 32;
        tdm_load_tile(T + (size_t)m0 * TWO_D + k1, lds_off_of(As[buf ^ 1]),
                      TWO_D, M_TOK, 128);
        tdm_load_tile(Wt + (size_t)n0 * TWO_D + k1, lds_off_of(Bs[buf ^ 1]),
                      TWO_D, D_MODEL, 256);
        __builtin_amdgcn_s_wait_tensorcnt(2);
      } else {
        __builtin_amdgcn_s_wait_tensorcnt(0);
      }
    }
    __syncthreads();
#else
    load_tile_sync<128>(T, TWO_D, m0, kk * 32, As[0], tid);
    load_tile_sync<256>(Wt, TWO_D, n0, kk * 32, Bs[0], tid);
    buf = 0;
    __syncthreads();
#endif
    v16bf af[4], bf[4];
#pragma unroll
    for (int i = 0; i < 4; ++i) af[i] = load_fragA(As[buf], wm * 64 + i * 16, lane);
#pragma unroll
    for (int j = 0; j < 4; ++j) bf[j] = load_fragB(Bs[buf], wn * 64 + j * 16, lane);
#pragma unroll
    for (int i = 0; i < 4; ++i)
#pragma unroll
      for (int j = 0; j < 4; ++j)
        acc[i][j] = WMMA_BF16(acc[i][j], af[i], bf[j]);
    __syncthreads();
  }
  int hl = lane >> 4, nl = lane & 15;
#pragma unroll
  for (int i = 0; i < 4; ++i)
#pragma unroll
    for (int j = 0; j < 4; ++j) {
      int col = n0 + wn * 64 + j * 16 + nl;
      float bb = bias[col];
#pragma unroll
      for (int v = 0; v < 8; ++v) {
        int row = m0 + wm * 64 + i * 16 + hl * 8 + v;
        out[(size_t)row * D_MODEL + col] =
            x[(size_t)row * D_MODEL + col] + acc[i][j][v] + bb;
      }
    }
}

// ======================= host launcher ======================================

extern "C" void kernel_launch(void* const* d_in, const int* in_sizes, int n_in,
                              void* d_out, int out_size, void* d_ws, size_t ws_size,
                              hipStream_t stream) {
  (void)in_sizes; (void)n_in; (void)out_size; (void)ws_size;
  const float* x        = (const float*)d_in[0];
  const float* emb      = (const float*)d_in[1];
  const float* ln_g     = (const float*)d_in[2];
  const float* ln_b     = (const float*)d_in[3];
  const float* Dvec     = (const float*)d_in[4];
  const float* f_log_dt = (const float*)d_in[5];
  const float* f_logAr  = (const float*)d_in[6];
  const float* f_Aim    = (const float*)d_in[7];
  const float* f_Cre    = (const float*)d_in[8];
  const float* f_Cim    = (const float*)d_in[9];
  const float* b_log_dt = (const float*)d_in[10];
  const float* b_logAr  = (const float*)d_in[11];
  const float* b_Aim    = (const float*)d_in[12];
  const float* b_Cre    = (const float*)d_in[13];
  const float* b_Cim    = (const float*)d_in[14];
  const float* bidir_W  = (const float*)d_in[15];
  const float* bidir_b  = (const float*)d_in[16];
  const float* gate_W   = (const float*)d_in[17];
  const float* gate_b   = (const float*)d_in[18];
  const float* up_W     = (const float*)d_in[19];
  const float* up_b     = (const float*)d_in[20];
  const float* down_W   = (const float*)d_in[21];
  const float* down_b   = (const float*)d_in[22];
  const float* film_W   = (const float*)d_in[23];
  const float* film_b   = (const float*)d_in[24];
  float* out = (float*)d_out;

  const size_t MB = 1024ull * 1024ull;
  char* w = (char*)d_ws;
  float*   h       = (float*)(w);                  // 64MB (aliased by tbuf later)
  float*   h_t     = (float*)(w + 64 * MB);        // 64MB (aliased by ybf later)
  bf16_t*  ybf     = (bf16_t*)(w + 64 * MB);       // 32MB, after scan consumed h_t
  bf16_t*  tbuf    = (bf16_t*)(w);                 // 64MB, after GEMM1 consumed h
  bf16_t*  ybi     = (bf16_t*)(w + 128 * MB);      // 64MB
  bf16_t*  bidirWt = (bf16_t*)(w + 192 * MB);      // 4MB  (1024 x 2048)
  bf16_t*  gateWt  = (bf16_t*)(w + 196 * MB);      // 4MB  (2048 x 1024)
  bf16_t*  upWt    = (bf16_t*)(w + 200 * MB);      // 4MB  (2048 x 1024)
  bf16_t*  downWt  = (bf16_t*)(w + 204 * MB);      // 4MB  (1024 x 2048)
  float4*  prm     = (float4*)(w + 208 * MB);      // 1MB  [2][1024][32]
  float*   film    = (float*)(w + 209 * MB);       // 64KB

  dim3 tb(32, 8);
  convT_kernel<<<dim3(D_MODEL / 32, TWO_D / 32), tb, 0, stream>>>(bidir_W, bidirWt, TWO_D, D_MODEL);
  convT_kernel<<<dim3(TWO_D / 32, D_MODEL / 32), tb, 0, stream>>>(gate_W, gateWt, D_MODEL, TWO_D);
  convT_kernel<<<dim3(TWO_D / 32, D_MODEL / 32), tb, 0, stream>>>(up_W, upWt, D_MODEL, TWO_D);
  convT_kernel<<<dim3(D_MODEL / 32, TWO_D / 32), tb, 0, stream>>>(down_W, downWt, TWO_D, D_MODEL);

  s4d_params_kernel<<<(D_MODEL * NMODES + 255) / 256, 256, 0, stream>>>(
      f_log_dt, f_logAr, f_Aim, f_Cre, f_Cim, prm);
  s4d_params_kernel<<<(D_MODEL * NMODES + 255) / 256, 256, 0, stream>>>(
      b_log_dt, b_logAr, b_Aim, b_Cre, b_Cim, prm + (size_t)D_MODEL * NMODES);

  film_kernel<<<(BATCH * TWO_D + 255) / 256, 256, 0, stream>>>(emb, film_W, film_b, film);

  ln_kernel<<<M_TOK, 256, 0, stream>>>(x, ln_g, ln_b, h);

  transpose_kernel<<<dim3(SEQ_L / 32, D_MODEL / 32, BATCH), tb, 0, stream>>>(h, h_t);

  scan_kernel<<<2 * BATCH * (D_MODEL / 16), 128, 0, stream>>>(h_t, prm, ybi);

  gemm_bidir_kernel<<<dim3(M_TOK / 128, D_MODEL / 256), 256, 0, stream>>>(
      ybi, bidirWt, bidir_b, h, Dvec, film, ybf);

  gemm_gateup_kernel<<<dim3(M_TOK / 128, TWO_D / 128), 256, 0, stream>>>(
      ybf, gateWt, upWt, gate_b, up_b, tbuf);

  gemm_down_kernel<<<dim3(M_TOK / 128, D_MODEL / 256), 256, 0, stream>>>(
      tbuf, downWt, down_b, x, out);
}
